// RNNCell_32435593019447
// MI455X (gfx1250) — matrix operand order repaired and checked by reading, not compiled, not run on hardware
//
#include <hip/hip_runtime.h>
#include <hip/hip_bf16.h>
#include <math.h>

// ---------------------------------------------------------------------------
// RNN: y,h_last = scan(tanh(wx_t + h@Whh^T + b)),  wx = x@Wih^T + b_ih
// B=64, S=1024, I=H=1024.  bf16 WMMA (v_wmma_f32_16x16x32_bf16), f32 accum.
//
// Phase 1: wx GEMM; x tiles staged to LDS by the Tensor Data Mover
//          (tensor_load_to_lds, TENSORcnt), double-buffered; NT stores to y.
// Phase 2: ONE persistent kernel; batch rows are independent across the scan,
//          so each block owns 16 batch rows and keeps h in LDS for all
//          1024 steps (sync = __syncthreads only).  h0 staged via
//          global_load_async_to_lds_b128 (ASYNCcnt).
// ---------------------------------------------------------------------------

typedef __attribute__((ext_vector_type(16))) __bf16   v16bf;
typedef __attribute__((ext_vector_type(8)))  __bf16   v8bf;
typedef __attribute__((ext_vector_type(8)))  float    v8f;
typedef __attribute__((ext_vector_type(4)))  int      v4i;
typedef __attribute__((ext_vector_type(8)))  int      v8i;
typedef __attribute__((ext_vector_type(4)))  unsigned u32x4;
typedef __attribute__((address_space(3)))    void     as3void;

#if defined(__has_builtin)
#if __has_builtin(__builtin_amdgcn_global_load_async_to_lds_b128) && \
    __has_builtin(__builtin_amdgcn_s_wait_asynccnt)
#define USE_ASYNC_LDS 1
#endif
#if __has_builtin(__builtin_amdgcn_tensor_load_to_lds) && \
    __has_builtin(__builtin_amdgcn_s_wait_tensorcnt)
#define USE_TDM 1
#endif
#endif
#ifndef USE_ASYNC_LDS
#define USE_ASYNC_LDS 0
#endif
#ifndef USE_TDM
#define USE_TDM 0
#endif

__device__ __forceinline__ void async_copy_b128(const void* gsrc, void* ldst) {
#if USE_ASYNC_LDS
  __builtin_amdgcn_global_load_async_to_lds_b128((v4i*)gsrc, (v4i*)ldst, 0, 0);
#else
  *(float4*)ldst = *(const float4*)gsrc;
#endif
}

__device__ __forceinline__ void async_wait_all() {
#if USE_ASYNC_LDS
  __builtin_amdgcn_s_wait_asynccnt(0);
#endif
}

// LDS byte offset of a __shared__ object (generic -> addrspace(3) -> int)
__device__ __forceinline__ unsigned lds_offset(const void* p) {
  return (unsigned)(unsigned long long)(as3void*)p;
}

#if USE_TDM
// Issue one TDM 2D tile load: tile1 rows x tile0 elems, element = 1<<ds_log2
// bytes, row stride stride0 elems.  D# layout per CDNA5 ISA ch.8.3/8.4.
__device__ __forceinline__ void tdm_load_2d(const void* gsrc, unsigned lds_off,
                                            unsigned ds_log2,
                                            unsigned tile0, unsigned tile1,
                                            unsigned td0, unsigned td1,
                                            unsigned stride0) {
  unsigned long long ga = (unsigned long long)gsrc;
  u32x4 g0;
  g0.x = 1u;                                               // count=1 (valid)
  g0.y = lds_off;                                          // lds_addr (bytes)
  g0.z = (unsigned)ga;                                     // global_addr lo
  g0.w = (unsigned)((ga >> 32) & 0x01FFFFFFull) | (2u << 30);  // type=2
  v8i g1;
  g1[0] = (int)(ds_log2 << 16);                            // wg_mask=0,data_size
  g1[1] = (int)((td0 & 0xFFFFu) << 16);                    // tensor_dim0[15:0]
  g1[2] = (int)((td0 >> 16) | ((td1 & 0xFFFFu) << 16));    // td0 hi | td1 lo
  g1[3] = (int)(((td1 >> 16) & 0xFFFFu) | (tile0 << 16));  // td1 hi | tile_dim0
  g1[4] = (int)tile1;                                      // tile_dim1, dim2=0
  g1[5] = (int)stride0;                                    // dim0_stride[31:0]
  g1[6] = 0;
  g1[7] = 0;
  v4i g2 = {0, 0, 0, 0};
  v4i g3 = {0, 0, 0, 0};
#if __clang_major__ >= 23
  v8i g4 = {0, 0, 0, 0, 0, 0, 0, 0};
  __builtin_amdgcn_tensor_load_to_lds(g0, g1, g2, g3, g4, 0);
#else
  __builtin_amdgcn_tensor_load_to_lds(g0, g1, g2, g3, 0);
#endif
}
#endif

__device__ __forceinline__ v8f wmma_bf16(v16bf a, v16bf b, v8f c) {
  // 8 args: (neg_a, A, neg_b, B, c_mod, C, reuse_a, reuse_b)
  return __builtin_amdgcn_wmma_f32_16x16x32_bf16(
      false, a, false, b, (short)0, c, false, false);
}

// B fragment: W row-major [N][K] (bf16). lane covers n, runs
// K = k0+8*half+[0..8) and K = k0+8*half+16+[0..8) -> two 16B loads per lane.
__device__ __forceinline__ v16bf load_b_frag(const __bf16* __restrict__ W,
                                             int ldw, int n, int k0, int half) {
  const __bf16* p = W + (size_t)n * ldw + k0 + 8 * half;
  union { v16bf v; v8bf h[2]; } u;
  u.h[0] = *(const v8bf*)(p);
  u.h[1] = *(const v8bf*)(p + 16);
  return u.v;
}

// A fragment from bf16 LDS tile (row-major, leading dim ld):
// lane row m, runs K = k0+8*half+[0..8) and K = k0+8*half+16+[0..8)
__device__ __forceinline__ v16bf load_a_lds(const __bf16* hs, int ld,
                                            int m, int k0, int half) {
  const __bf16* p = hs + m * ld + k0 + 8 * half;
  union { v16bf v; v8bf h[2]; } u;
  u.h[0] = *(const v8bf*)(p);
  u.h[1] = *(const v8bf*)(p + 16);
  return u.v;
}

// A fragment from f32 data (global or LDS), converted to bf16 in registers.
__device__ __forceinline__ v16bf cvt_a_f32(const float* p) {
  float4 f0 = *(const float4*)(p);
  float4 f1 = *(const float4*)(p + 4);
  float4 f2 = *(const float4*)(p + 16);
  float4 f3 = *(const float4*)(p + 20);
  v16bf a;
  a[0]  = (__bf16)f0.x; a[1]  = (__bf16)f0.y; a[2]  = (__bf16)f0.z; a[3]  = (__bf16)f0.w;
  a[4]  = (__bf16)f1.x; a[5]  = (__bf16)f1.y; a[6]  = (__bf16)f1.z; a[7]  = (__bf16)f1.w;
  a[8]  = (__bf16)f2.x; a[9]  = (__bf16)f2.y; a[10] = (__bf16)f2.z; a[11] = (__bf16)f2.w;
  a[12] = (__bf16)f3.x; a[13] = (__bf16)f3.y; a[14] = (__bf16)f3.z; a[15] = (__bf16)f3.w;
  return a;
}

// --------------------------- weight conversion -----------------------------
__global__ void convert_kernel(const float* __restrict__ Wih,
                               const float* __restrict__ Whh,
                               __bf16* __restrict__ W1,
                               __bf16* __restrict__ W2, int nW) {
  int i = blockIdx.x * blockDim.x + threadIdx.x;
  int stride = gridDim.x * blockDim.x;
  for (int j = i; j < nW; j += stride) {
    W1[j] = (__bf16)Wih[j];
    W2[j] = (__bf16)Whh[j];
  }
}

// ------------------------ phase 1: wx = x @ Wih^T + b ----------------------
// grid (M/64, N/128), 256 threads = 8 waves (2M x 4N); each wave 32x32 via
// 2x2 WMMA fragments.  x tile [64 x 128] f32 double-buffered in LDS, staged
// by the Tensor Data Mover (fallback: async-to-LDS / plain copies).
__global__ void gemm_wx_kernel(const float* __restrict__ X,
                               const __bf16* __restrict__ W1,
                               const float* __restrict__ bias,
                               float* __restrict__ Y) {
  const int K = 1024, N = 1024, KC = 128;
  __shared__ float xs[2][64 * KC];                 // 2 x 32 KB

  int tid = threadIdx.x;
  int lane = tid & 31;
  int wave = tid >> 5;
  int half = lane >> 4;
  int wm = wave >> 2;                              // 0..1
  int wn = wave & 3;                               // 0..3
  size_t mBase = (size_t)blockIdx.x * 64;
  int nBase = blockIdx.y * 128 + wn * 32;

  // stage x[mBase:+64, k0:+128) f32 into xs[buf]
  auto stage = [&](int buf, int k0) {
#if USE_TDM
    if (wave == 0)                                 // one DMA per stage
      tdm_load_2d(X + mBase * K + k0, lds_offset(&xs[buf][0]),
                  /*4B elems*/ 2u, /*tile0*/ KC, /*tile1*/ 64,
                  /*td0*/ (unsigned)K, /*td1*/ 65536u, /*stride0*/ (unsigned)K);
#else
#pragma unroll
    for (int i = 0; i < 8; i++) {
      int j = (tid + 256 * i) * 4;                 // float index in tile
      int row = j >> 7;
      int col = j & 127;
      const float* src = X + (mBase + row) * K + k0 + col;
      async_copy_b128(src, &xs[buf][j]);
    }
#endif
  };
  auto stage_wait = [&]() {
#if USE_TDM
    if (wave == 0) __builtin_amdgcn_s_wait_tensorcnt(0);
#else
    async_wait_all();
#endif
    __syncthreads();
  };

  stage(0, 0);
  stage_wait();

  v8f c00 = {}, c01 = {}, c10 = {}, c11 = {};
  int m0 = wm * 32 + (lane & 15);                  // local A row (tile 0)

  for (int kc = 0; kc < K; kc += KC) {
    int cur = (kc / KC) & 1;
    if (kc + KC < K) stage(cur ^ 1, kc + KC);      // overlap DMA with math
#pragma unroll
    for (int kk = 0; kk < KC; kk += 32) {
      v16bf a0 = cvt_a_f32(&xs[cur][m0 * KC + kk + 8 * half]);
      v16bf a1 = cvt_a_f32(&xs[cur][(m0 + 16) * KC + kk + 8 * half]);
      v16bf b0 = load_b_frag(W1, K, nBase + (lane & 15), kc + kk, half);
      v16bf b1 = load_b_frag(W1, K, nBase + 16 + (lane & 15), kc + kk, half);
      c00 = wmma_bf16(a0, b0, c00);
      c01 = wmma_bf16(a0, b1, c01);
      c10 = wmma_bf16(a1, b0, c10);
      c11 = wmma_bf16(a1, b1, c11);
    }
    stage_wait();
  }

  // epilogue: +bias, stream to y non-temporally (read once much later)
  int n0 = nBase + (lane & 15);
  float bn0 = bias[n0];
  float bn1 = bias[n0 + 16];
#pragma unroll
  for (int r = 0; r < 8; r++) {
    size_t m = mBase + wm * 32 + r + 8 * half;
    __builtin_nontemporal_store(c00[r] + bn0, &Y[m * N + n0]);
    __builtin_nontemporal_store(c01[r] + bn1, &Y[m * N + n0 + 16]);
    __builtin_nontemporal_store(c10[r] + bn0, &Y[(m + 16) * N + n0]);
    __builtin_nontemporal_store(c11[r] + bn1, &Y[(m + 16) * N + n0 + 16]);
  }
}

// ---------------- phase 2: persistent scan over S timesteps ----------------
// Batch rows are independent across the recurrence, so block b owns rows
// [16b, 16b+16) and keeps h resident in LDS for the whole scan.
// 1024 threads = 32 waves; wave w computes cols [32w, 32w+32) (16x32 tile).
__global__ void rnn_scan_kernel(const float* __restrict__ h0,
                                const __bf16* __restrict__ W2,
                                const float* __restrict__ bias,
                                float* __restrict__ Y,
                                float* __restrict__ hlast) {
  const int H = 1024, S = 1024;
  __shared__ __bf16 hs[16 * 1024];                 // 32 KB resident h tile
  __shared__ float  htmp[4 * 1024];                // 16 KB async staging

  int tid = threadIdx.x;
  int lane = tid & 31;
  int wave = tid >> 5;                             // 0..31
  int half = lane >> 4;
  int mrow = blockIdx.x * 16;                      // batch-row base
  int n0 = wave * 32 + (lane & 15);

  // initial h: async-copy f32 rows to LDS, convert to resident bf16 tile
  for (int rr = 0; rr < 16; rr += 4) {
    int j = tid * 4;                               // one b128 per thread
    int row = j >> 10;
    int col = j & 1023;
    async_copy_b128(h0 + (size_t)(mrow + rr + row) * H + col, &htmp[j]);
    async_wait_all();
    __syncthreads();
#pragma unroll
    for (int c = 0; c < 4; c++)
      hs[(rr + row) * 1024 + col + c] = (__bf16)htmp[j + c];
    __syncthreads();
  }

  float bn0 = bias[n0];
  float bn1 = bias[n0 + 16];

  for (int t = 0; t < S; t++) {
    v8f c0 = {}, c1 = {};
#pragma unroll 4
    for (int k0 = 0; k0 < H; k0 += 32) {
      if (k0 + 128 < H)                            // Whh stream hint (L2-hot)
        __builtin_prefetch(W2 + (size_t)n0 * H + k0 + 128, 0, 0);
      v16bf a  = load_a_lds(hs, 1024, (lane & 15), k0, half);
      v16bf b0 = load_b_frag(W2, H, n0, k0, half);
      v16bf b1 = load_b_frag(W2, H, n0 + 16, k0, half);
      c0 = wmma_bf16(a, b0, c0);
      c1 = wmma_bf16(a, b1, c1);
    }
    __syncthreads();                               // all hs reads complete

    // h = tanh(wx + h@Whh^T + b); wx read in-place from Y and overwritten
#pragma unroll
    for (int r = 0; r < 8; r++) {
      int ml = r + 8 * half;                       // local row 0..15
      int m = mrow + ml;
      size_t yi = ((size_t)m * S + t) * H + n0;
      float v0 = tanhf(c0[r] + __builtin_nontemporal_load(&Y[yi]) + bn0);
      __builtin_nontemporal_store(v0, &Y[yi]);
      hs[ml * 1024 + n0] = (__bf16)v0;
      float v1 = tanhf(c1[r] + __builtin_nontemporal_load(&Y[yi + 16]) + bn1);
      __builtin_nontemporal_store(v1, &Y[yi + 16]);
      hs[ml * 1024 + n0 + 16] = (__bf16)v1;
      if (t == S - 1) {
        hlast[m * H + n0] = v0;
        hlast[m * H + n0 + 16] = v1;
      }
    }
    __syncthreads();                               // hs updated for next step
  }
}

// ---------------------------------------------------------------------------
extern "C" void kernel_launch(void* const* d_in, const int* in_sizes, int n_in,
                              void* d_out, int out_size, void* d_ws, size_t ws_size,
                              hipStream_t stream) {
  (void)in_sizes; (void)n_in; (void)out_size; (void)ws_size;
  const int B = 64, S = 1024, I = 1024, H = 1024;

  const float* x     = (const float*)d_in[0];
  const float* h0    = (const float*)d_in[1];
  const float* Wih_w = (const float*)d_in[2];
  const float* Wih_b = (const float*)d_in[3];
  const float* Whh_w = (const float*)d_in[4];
  const float* Whh_b = (const float*)d_in[5];

  float* y     = (float*)d_out;                    // [B,S,H]
  float* hlast = y + (size_t)B * S * H;            // [B,H]

  __bf16* W1 = (__bf16*)d_ws;                      // Wih bf16 [H][I]
  __bf16* W2 = W1 + (size_t)H * I;                 // Whh bf16 [H][H]

  convert_kernel<<<2048, 256, 0, stream>>>(Wih_w, Whh_w, W1, W2, H * I);

  dim3 g1((B * S) / 64, H / 128);
  gemm_wx_kernel<<<g1, 256, 0, stream>>>(x, W1, Wih_b, y);

  rnn_scan_kernel<<<B / 16, 1024, 0, stream>>>(h0, W2, Whh_b, y, hlast);
}
